// minifloatLinearPTQ_14860586844384
// MI455X (gfx1250) — 8-bit matrix operand order repaired and checked by reading, compiled, not run on hardware
//
#include <hip/hip_runtime.h>

typedef __attribute__((ext_vector_type(16))) int   v16i;
typedef __attribute__((ext_vector_type(8)))  float v8f;
typedef __attribute__((ext_vector_type(4)))  int   v4i;
typedef __attribute__((ext_vector_type(4)))  float v4f;

#define KDIM 256
#define NDIM 256
#define ROWS_PER_WG 128
#define W_BYTES (NDIM * KDIM)

__global__ void quantize_wb_kernel(const float* __restrict__ w,
                                   const float* __restrict__ b,
                                   unsigned int* __restrict__ wq,
                                   float* __restrict__ bq) {
  int tid = blockIdx.x * blockDim.x + threadIdx.x;
  if (tid < W_BYTES / 4) {
    v4f f = ((const v4f*)w)[tid];
    int p = 0;
    p = __builtin_amdgcn_cvt_pk_fp8_f32(f.x, f.y, p, false);
    p = __builtin_amdgcn_cvt_pk_fp8_f32(f.z, f.w, p, true);
    wq[tid] = (unsigned int)p;
  } else if (tid < W_BYTES / 4 + NDIM) {
    int i = tid - W_BYTES / 4;
    int p = __builtin_amdgcn_cvt_pk_fp8_f32(b[i], 0.0f, 0, false);
    bq[i] = __builtin_amdgcn_cvt_f32_fp8(p, 0);
  }
}

__global__ __launch_bounds__(256) void mf_linear_kernel(
    const float* __restrict__ x,
    const unsigned char* __restrict__ wq,
    const float* __restrict__ bq,
    float* __restrict__ out) {
  __shared__ unsigned char wlds[W_BYTES];

  const int t = threadIdx.x;

  {
    const v4i* src = (const v4i*)(wq + t * KDIM);
    #pragma unroll
    for (int i = 0; i < 16; ++i) {
      int chunk = i ^ (t & 15);
      *(v4i*)(&wlds[t * KDIM + chunk * 16]) = src[i];
    }
  }

  const int wave = t >> 5;
  const int lane = t & 31;
  const int m    = lane & 15;
  const int hi   = lane >> 4;

  const size_t rowbase = (size_t)blockIdx.x * ROWS_PER_WG + (size_t)wave * 16;
  const float* xrow = x + (rowbase + (size_t)m) * KDIM;

  v16i a0, a1;
  #pragma unroll
  for (int v = 0; v < 16; ++v) {
    const int klocal = ((v >> 2) * 32) + hi * 16 +
                       ((v & 3) * 4);
    {
      v4f f = *(const v4f*)(xrow + klocal);
      int p = 0;
      p = __builtin_amdgcn_cvt_pk_fp8_f32(f.x, f.y, p, false);
      p = __builtin_amdgcn_cvt_pk_fp8_f32(f.z, f.w, p, true);
      a0[v] = p;
    }
    {
      v4f f = *(const v4f*)(xrow + 128 + klocal);
      int p = 0;
      p = __builtin_amdgcn_cvt_pk_fp8_f32(f.x, f.y, p, false);
      p = __builtin_amdgcn_cvt_pk_fp8_f32(f.z, f.w, p, true);
      a1[v] = p;
    }
  }

  __syncthreads();

  #pragma unroll
  for (int nt = 0; nt < 16; ++nt) {
    const int wrow = nt * 16 + m;
    v8f c = {};

    #pragma unroll
    for (int kb = 0; kb < 2; ++kb) {
      v16i bt;
      #pragma unroll
      for (int g = 0; g < 4; ++g) {
        int chunk = (kb * 8 + g * 2 + hi) ^ (wrow & 15);
        v4i q = *(const v4i*)(&wlds[wrow * KDIM + chunk * 16]);
        bt[4 * g + 0] = q.x;
        bt[4 * g + 1] = q.y;
        bt[4 * g + 2] = q.z;
        bt[4 * g + 3] = q.w;
      }
      c = __builtin_amdgcn_wmma_f32_16x16x128_fp8_fp8(
              (kb == 0) ? a0 : a1, bt, (short)0, c,
               false,  false);
    }

    const float bias = bq[wrow];
    float* orow = out + (rowbase + (size_t)hi * 8) * NDIM + wrow;
    #pragma unroll
    for (int r = 0; r < 8; ++r) {
      orow[(size_t)r * NDIM] = c[r] + bias;
    }
  }
}

extern "C" void kernel_launch(void* const* d_in, const int* in_sizes, int n_in,
                              void* d_out, int out_size, void* d_ws, size_t ws_size,
                              hipStream_t stream) {
  const float* x = (const float*)d_in[0];
  const float* w = (const float*)d_in[1];
  const float* b = (const float*)d_in[2];

  unsigned int* wq_u32 = (unsigned int*)d_ws;
  float*        bq     = (float*)((char*)d_ws + W_BYTES);

  {
    int threads = W_BYTES / 4 + NDIM;
    int blocks  = (threads + 255) / 256;
    quantize_wb_kernel<<<blocks, 256, 0, stream>>>(w, b, wq_u32, bq);
  }

  {
    int tokens = in_sizes[0] / KDIM;
    int blocks = tokens / ROWS_PER_WG;
    mf_linear_kernel<<<blocks, 256, 0, stream>>>(
        x, (const unsigned char*)d_ws, bq, (float*)d_out);
  }
}
